// resblock_77464030151164
// MI455X (gfx1250) — matrix operand order repaired and checked by reading, not compiled, not run on hardware
//
#include <hip/hip_runtime.h>

#define NPTS 200000
#define KOFF 27
#define CH   128
#define BM   128               // rows per workgroup
#define TILE_ELE (BM * CH)     // f16 elements per LDS tile buffer (32 KB)

typedef __attribute__((ext_vector_type(16))) _Float16 v16h;
typedef __attribute__((ext_vector_type(8)))  float    v8f;
typedef __attribute__((ext_vector_type(4)))  unsigned int v4u;
typedef __attribute__((ext_vector_type(4)))  int       v4i;

union FragU { v4u q[2]; v16h h; };

#ifndef __has_builtin
#define __has_builtin(x) 0
#endif

#if __has_builtin(__builtin_amdgcn_global_load_async_to_lds_b128)
#define USE_ASYNC_LDS 1
#else
#define USE_ASYNC_LDS 0
#endif

#if __has_builtin(__builtin_amdgcn_s_wait_asynccnt)
#define WAIT_ASYNC(n) __builtin_amdgcn_s_wait_asynccnt(n)
#else
#define WAIT_ASYNC(n) asm volatile("s_wait_asynccnt %0" ::"n"(n) : "memory")
#endif

// address-space-qualified int4 pointers for the async-to-LDS builtin
typedef __attribute__((address_space(1))) v4i as1_v4i;
typedef __attribute__((address_space(3))) v4i as3_v4i;

__device__ __forceinline__ v8f splat8(float v) {
  v8f r;
#pragma unroll
  for (int i = 0; i < 8; ++i) r[i] = v;
  return r;
}

// ---------- prep: f32 -> f16 convert (4 elems / thread) ----------
__global__ __launch_bounds__(256) void cvt_f32_to_f16(const float* __restrict__ s,
                                                      _Float16* __restrict__ d, int n4) {
  int i = blockIdx.x * 256 + threadIdx.x;
  if (i >= n4) return;
  float4 v = ((const float4*)s)[i];
  alignas(8) _Float16 h[4] = {(_Float16)v.x, (_Float16)v.y, (_Float16)v.z, (_Float16)v.w};
  ((uint2*)d)[i] = *(const uint2*)h;
}

// ---------- prep: pack W[k][cin][cout] into per-lane WMMA B fragments ----------
// Fragment f = (k*4 + ks)*8 + nt holds B (32x16) for K-step ks, N-tile nt.
// B 32x16 lane layout: lane<16 -> Klocal 0..7,16..23, lane>=16 -> Klocal 8..15,24..31;
// col = nt*16 + (lane&15); v4u q holds Klocal 16q + 8*(lane>=16) + 0..7. Two contiguous v4u per lane.
__global__ __launch_bounds__(256) void pack_w_frags(const float* __restrict__ w,
                                                    v4u* __restrict__ wf) {
  int t = blockIdx.x * 256 + threadIdx.x;
  if (t >= KOFF * 4 * 8 * 64) return;
  int lane = t & 31;
  int q    = (t >> 5) & 1;
  int f    = t >> 6;
  int nt   = f & 7;
  int ks   = (f >> 3) & 3;
  int k    = f >> 5;
  int col  = nt * 16 + (lane & 15);
  int k0   = ks * 32 + ((lane & 16) ? 8 : 0) + q * 16;
  alignas(16) _Float16 h[8];
#pragma unroll
  for (int j = 0; j < 8; ++j)
    h[j] = (_Float16)w[((size_t)k * CH + (size_t)(k0 + j)) * CH + col];
  wf[(size_t)f * 64 + (size_t)lane * 2 + q] = *(const v4u*)h;
}

// ---------- main: one sparse-conv layer via WMMA ----------
// LAYER 0: hout = f16(relu(conv(xin,W)+b))
// LAYER 1: fout = relu(conv(xin,W)+b) + skip
template <int LAYER>
__global__ __launch_bounds__(256) void spconv_wmma(
    const _Float16* __restrict__ xin,      // [NPTS, CH] f16 features to gather
    const int* __restrict__ nbr,           // [NPTS, KOFF]
    const unsigned char* __restrict__ msk, // [NPTS, KOFF] bool
    const v4u* __restrict__ wf,            // packed B fragments
    const float* __restrict__ bias,        // [CH]
    const float* __restrict__ skip,        // [NPTS, CH] f32 (layer 1 only)
    const _Float16* __restrict__ zrow,     // [CH] zeros (masked-off gather target)
    _Float16* __restrict__ hout,           // layer-0 output (f16)
    float* __restrict__ fout)              // layer-1 output (f32)
{
  // double-buffered tile, stride CH, XOR chunk swizzle (chunk' = chunk ^ (row&15))
  __shared__ _Float16 Als[2 * TILE_ELE];

  const int tid  = threadIdx.x;
  const int n0   = blockIdx.x * BM;
  const int wave = tid >> 5;
  const int lane = tid & 31;

  // gather role: 2 threads per row, each copies 128B (8 x b128)
  const int grow  = tid >> 1;
  const int ghalf = tid & 1;
  const int gn    = n0 + grow;
  const bool gvalid = (gn < NPTS);
  const int gsw = grow & 15;

  // resolve gather source row-half (zero row when masked off / out of range)
  auto srcrow = [&](int kk) -> const _Float16* {
    const _Float16* base = zrow;
    if (gvalid) {
      const unsigned char m = msk[(size_t)gn * KOFF + kk];
      if (m) {
        const int gi = nbr[(size_t)gn * KOFF + kk];
        base = xin + (size_t)gi * CH;
      }
    }
    return base + ghalf * 64;
  };

  // accumulators: 8 N-tiles of 16x16 f32, bias-initialized
  v8f acc[8];
#pragma unroll
  for (int nt = 0; nt < 8; ++nt)
    acc[nt] = splat8(bias[nt * 16 + (lane & 15)]);

  // A-fragment addressing (16-bit A 16x32): row = mrow + lane%16;
  // lane>=16 takes +8 K-chunk; chunks (8 f16) XOR-swizzled by row&15
  const int arow = lane & 15;
  const int ahi  = (lane >> 4) & 1;
  _Float16* const arbase = Als + (wave * 16 + arow) * CH;

  auto compute = [&](int k) {
    const _Float16* ab = arbase + (k & 1) * TILE_ELE;
#pragma unroll
    for (int ks = 0; ks < 4; ++ks) {
      const int c0 = ks * 4 + ahi;
      FragU A;
      A.q[0] = *(const v4u*)(ab + ((c0 ^ arow) * 8));
      A.q[1] = *(const v4u*)(ab + (((c0 + 2) ^ arow) * 8));
      const v4u* bptr = wf + (size_t)((k * 4 + ks) * 8) * 64 + (size_t)lane * 2;
#pragma unroll
      for (int nt = 0; nt < 8; ++nt) {
        FragU B;
        B.q[0] = bptr[nt * 64];
        B.q[1] = bptr[nt * 64 + 1];
        acc[nt] = __builtin_amdgcn_wmma_f32_16x16x32_f16(
            false, A.h, false, B.h, (short)0, acc[nt], false, false);
      }
    }
  };

#if USE_ASYNC_LDS
  // async gather: 8 x global_load_async_to_lds_b128 per thread per tile
  auto issue = [&](int kk) {
    const _Float16* s = srcrow(kk);
#pragma unroll
    for (int j = 0; j < 8; ++j) {
      _Float16* d = Als + (kk & 1) * TILE_ELE + grow * CH +
                    (((ghalf * 8 + j) ^ gsw) * 8);
      __builtin_amdgcn_global_load_async_to_lds_b128(
          (as1_v4i*)(s + j * 8), (as3_v4i*)d, 0, 0);
    }
  };

  issue(0);
  for (int k = 0; k < KOFF; ++k) {
    __syncthreads();                 // all waves done with compute(k-1)
    if (k + 1 < KOFF) {
      issue(k + 1);                  // overlaps with compute(k)
      WAIT_ASYNC(8);                 // tile-k's 8 async ops (oldest) done
    } else {
      WAIT_ASYNC(0);
    }
    __syncthreads();                 // tile k fully landed for all threads
    compute(k);
  }
#else
  // sync fallback: prefetch into regs, store to LDS, double-buffered
  v4u pre[8];
  auto gload = [&](int kk) {
    const v4u* s = (const v4u*)srcrow(kk);
#pragma unroll
    for (int j = 0; j < 8; ++j) pre[j] = s[j];
  };
  auto lstore = [&](int kk) {
#pragma unroll
    for (int j = 0; j < 8; ++j)
      *(v4u*)(Als + (kk & 1) * TILE_ELE + grow * CH +
              (((ghalf * 8 + j) ^ gsw) * 8)) = pre[j];
  };

  gload(0);
  for (int k = 0; k < KOFF; ++k) {
    __syncthreads();
    lstore(k);
    if (k + 1 < KOFF) gload(k + 1);  // overlaps with compute(k)
    __syncthreads();
    compute(k);
  }
#endif

  // ---- epilogue: ReLU (+skip) and store; C/D layout: VGPR v -> row v(+8), lane%16 -> col
  const int rbase = wave * 16 + ((lane & 16) ? 8 : 0);
#pragma unroll
  for (int nt = 0; nt < 8; ++nt) {
    const int col = nt * 16 + (lane & 15);
#pragma unroll
    for (int v = 0; v < 8; ++v) {
      const int n = n0 + rbase + v;
      if (n < NPTS) {
        float val = fmaxf(acc[nt][v], 0.0f);
        if (LAYER == 0) {
          hout[(size_t)n * CH + col] = (_Float16)val;
        } else {
          fout[(size_t)n * CH + col] = val + skip[(size_t)n * CH + col];
        }
      }
    }
  }
}

extern "C" void kernel_launch(void* const* d_in, const int* in_sizes, int n_in,
                              void* d_out, int out_size, void* d_ws, size_t ws_size,
                              hipStream_t stream) {
  const float*         x    = (const float*)d_in[0];
  const int*           nbr  = (const int*)d_in[1];
  const unsigned char* msk  = (const unsigned char*)d_in[2];
  const float*         W1   = (const float*)d_in[3];
  const float*         b1   = (const float*)d_in[4];
  const float*         W2   = (const float*)d_in[5];
  const float*         b2   = (const float*)d_in[6];
  float*               out  = (float*)d_out;

  // workspace layout
  _Float16* xh   = (_Float16*)d_ws;                 // N*CH f16
  _Float16* hh   = xh + (size_t)NPTS * CH;          // N*CH f16
  v4u*      w1f  = (v4u*)(hh + (size_t)NPTS * CH);  // 864*64 v4u
  v4u*      w2f  = w1f + (size_t)KOFF * 4 * 8 * 64;
  _Float16* zrow = (_Float16*)(w2f + (size_t)KOFF * 4 * 8 * 64);  // CH f16 zeros

  (void)hipMemsetAsync(zrow, 0, CH * sizeof(_Float16), stream);

  // prep: x -> f16
  {
    int n4 = NPTS * CH / 4;
    cvt_f32_to_f16<<<(n4 + 255) / 256, 256, 0, stream>>>(x, xh, n4);
  }
  // prep: pack weights into WMMA B-fragment order
  {
    int tot = KOFF * 4 * 8 * 64;
    pack_w_frags<<<(tot + 255) / 256, 256, 0, stream>>>(W1, w1f);
    pack_w_frags<<<(tot + 255) / 256, 256, 0, stream>>>(W2, w2f);
  }

  const int nblk = (NPTS + BM - 1) / BM;
  // layer 1: h = relu(conv(x, W1) + b1)   (f16 intermediate, stays L2-resident)
  spconv_wmma<0><<<nblk, 256, 0, stream>>>(xh, nbr, msk, w1f, b1, nullptr, zrow, hh, nullptr);
  // layer 2: out = relu(conv(h, W2) + b2) + x
  spconv_wmma<1><<<nblk, 256, 0, stream>>>(hh, nbr, msk, w2f, b2, x, zrow, nullptr, out);
}
